// SensorTGNNBranch_14087492730977
// MI455X (gfx1250) — matrix operand order repaired and checked by reading, compiled, not run on hardware
//
#include <hip/hip_runtime.h>

// ---------------------------------------------------------------------------
// MI455X (gfx1250) fused banded-attention transformer.
//
// Roofline: 206 GFLOP of D=256 projections vs a 128 MB f32 activation tensor.
// At 23.3 TB/s HBM the workload is memory bound unless each layer is fused,
// so each workgroup processes a 64-row time tile entirely in 320KB WGP LDS:
//   LN -> QKV (v_wmma_f32_16x16x32_bf16) -> banded softmax -> O proj -> resid
// HBM traffic/layer = read h + write h = 256 MB. Weights (bf16, 1.5 MB) live
// in L2; the tile loop is structured so each wave reuses one set of B
// fragments across all 4 M-tiles (one pass over the weights per workgroup).
// ---------------------------------------------------------------------------

typedef __attribute__((ext_vector_type(16))) __bf16         v16bf;
typedef __attribute__((ext_vector_type(16))) unsigned short v16u;
typedef __attribute__((ext_vector_type(8)))  unsigned short u16x8;
typedef __attribute__((ext_vector_type(8)))  float          v8f;

#define DEV __device__ __forceinline__

static constexpr int B_     = 32;
static constexpr int T_     = 4096;
static constexpr int D_     = 256;
static constexpr int DH_    = 32;
static constexpr int DEPTH_ = 3;
static constexpr int TSZ    = 64;        // time rows per workgroup tile
static constexpr int NR     = TSZ + 2;   // + halo row each side
static constexpr int LP     = 264;       // padded LDS row stride (bf16 elems, 528B)
static constexpr long BT    = (long)B_ * T_;

// Native bf16 converts (single v_cvt / shift on gfx1250, RNE).
DEV unsigned short f2bf(float f) {
  __bf16 h = (__bf16)f;
  return __builtin_bit_cast(unsigned short, h);
}
DEV float bf2f(unsigned short u) {
  return (float)__builtin_bit_cast(__bf16, u);
}

// A fragment (16x32 bf16) from LDS, CDNA5 wave32 layout:
// lane m (0-15): K = kb+{0..7} then kb+{16..23}; lane m+16: +8 on both halves.
DEV v16bf load_frag_A(const unsigned short* base, int row0, int kb, int lane) {
  int m = lane & 15, half = lane >> 4;
  const unsigned short* p = base + (size_t)(row0 + m) * LP + kb + half * 8;
  u16x8 lo = *(const u16x8*)p;          // K = kb + half*8 + 0..7
  u16x8 hi = *(const u16x8*)(p + 16);   // K = kb + 16 + half*8 + 0..7
  v16u u;
#pragma unroll
  for (int i = 0; i < 8; ++i) { u[i] = lo[i]; u[i + 8] = hi[i]; }
  return __builtin_bit_cast(v16bf, u);
}

// B fragment (32x16 bf16) from global bf16 weights stored transposed [n][k]:
// lane l holds column n = nb + (l&15); K = kb + (l>>4)*8 + {0..7} then kb + 16 + (l>>4)*8 + {0..7}.
DEV v16bf load_frag_B(const unsigned short* wt, int nb, int kb, int lane) {
  int n = nb + (lane & 15), half = lane >> 4;
  const unsigned short* p = wt + (size_t)n * D_ + kb + half * 8;
  u16x8 lo = *(const u16x8*)p;
  u16x8 hi = *(const u16x8*)(p + 16);
  v16u u;
#pragma unroll
  for (int i = 0; i < 8; ++i) { u[i] = lo[i]; u[i + 8] = hi[i]; }
  return __builtin_bit_cast(v16bf, u);
}

DEV v8f wmma_bf16(v16bf a, v16bf b, v8f c) {
  return __builtin_amdgcn_wmma_f32_16x16x32_bf16(false, a, false, b,
                                                 (short)0, c, false, false);
}

// Store one 16x16 f32 accumulator tile to bf16 LDS (C/D layout: VGPR r ->
// row r + 8*(lane>>4), col = lane&15).
DEV void store_tile(unsigned short* dst, int row0, int n, int half, v8f acc) {
#pragma unroll
  for (int r = 0; r < 8; ++r)
    dst[(size_t)(row0 + r + 8 * half) * LP + n] = f2bf(acc[r]);
}

// ---------------------------------------------------------------------------
// One-time weight conversion: W[l][k][n] f32 -> wt[l][m][n][k] bf16 (transposed)
// ---------------------------------------------------------------------------
__global__ __launch_bounds__(256)
void convert_weights(const float* __restrict__ Wq, const float* __restrict__ Wk,
                     const float* __restrict__ Wv, const float* __restrict__ Wo,
                     unsigned short* __restrict__ wt) {
  int gid = blockIdx.x * blockDim.x + threadIdx.x;
  if (gid >= DEPTH_ * 4 * D_ * D_) return;
  int k = gid & 255;
  int n = (gid >> 8) & 255;
  int m = (gid >> 16) & 3;
  int l = gid >> 18;
  const float* src = (m == 0) ? Wq : (m == 1) ? Wk : (m == 2) ? Wv : Wo;
  wt[gid] = f2bf(src[(size_t)l * D_ * D_ + (size_t)k * D_ + n]);
}

// ---------------------------------------------------------------------------
// Input projection: h = s @ W_in + b_in  (K=3, pure bandwidth)
// ---------------------------------------------------------------------------
__global__ __launch_bounds__(256)
void in_proj(const float* __restrict__ s, const float* __restrict__ Win,
             const float* __restrict__ bin, float* __restrict__ h) {
  size_t row = blockIdx.x;
  int c = threadIdx.x;
  float s0 = s[row * 3 + 0], s1 = s[row * 3 + 1], s2 = s[row * 3 + 2];
  h[row * D_ + c] =
      fmaf(s0, Win[c], fmaf(s1, Win[D_ + c], fmaf(s2, Win[2 * D_ + c], bin[c])));
}

// ---------------------------------------------------------------------------
// Fused transformer layer: one workgroup = one 64-row time tile of one batch.
// ---------------------------------------------------------------------------
__global__ __launch_bounds__(256)
void layer_fused(const float* __restrict__ h_in, float* __restrict__ h_out,
                 const unsigned short* __restrict__ wt,   // [4][n=256][k=256] bf16
                 const float* __restrict__ bq, const float* __restrict__ bk,
                 const float* __restrict__ bv, const float* __restrict__ bo,
                 const float* __restrict__ lng, const float* __restrict__ lnb) {
  __shared__ __align__(16) unsigned short sh_hn[NR * LP];   // LN(h), rows t0-1..t0+64
  __shared__ __align__(16) unsigned short sh_q [TSZ * LP];  // rows t0..t0+63
  __shared__ __align__(16) unsigned short sh_k [NR * LP];   // rows t0-1..t0+64
  __shared__ __align__(16) unsigned short sh_v [NR * LP];
  __shared__ __align__(16) unsigned short sh_a [TSZ * LP];  // attention output

  const int b    = blockIdx.y;
  const int t0   = blockIdx.x * TSZ;
  const int lane = threadIdx.x & 31;
  const int wave = threadIdx.x >> 5;

  // Warm L2/WGP$ with the output-projection weights used last (Phase D).
  {
    const unsigned short* wo = wt + (size_t)3 * D_ * D_;
#pragma unroll
    for (int i = 0; i < 4; ++i)
      __builtin_prefetch(wo + ((size_t)threadIdx.x * 4 + i) * 64, 0, 1);
  }

  // ---- Phase A: LayerNorm rows [t0-1, t0+64] -> bf16 LDS (wave per row) ----
  for (int r = wave; r < NR; r += 8) {
    int t = t0 - 1 + r;
    if (t < 0 || t >= T_) {
#pragma unroll
      for (int j = 0; j < 8; ++j) sh_hn[r * LP + lane * 8 + j] = 0;
    } else {
      const float* row = h_in + ((size_t)b * T_ + t) * D_;
      float x[8], s = 0.f, s2 = 0.f;
#pragma unroll
      for (int j = 0; j < 8; ++j) {
        x[j] = row[lane * 8 + j];
        s += x[j]; s2 += x[j] * x[j];
      }
#pragma unroll
      for (int m = 16; m >= 1; m >>= 1) {
        s  += __shfl_xor(s,  m, 32);
        s2 += __shfl_xor(s2, m, 32);
      }
      float mu   = s * (1.f / D_);
      float rstd = rsqrtf(s2 * (1.f / D_) - mu * mu + 1e-5f);
#pragma unroll
      for (int j = 0; j < 8; ++j) {
        int c = lane * 8 + j;
        sh_hn[r * LP + c] = f2bf((x[j] - mu) * rstd * lng[c] + lnb[c]);
      }
    }
  }
  __syncthreads();

  // ---- Phase B: q/k/v = LN(h) @ W + b via WMMA bf16 (f32 accumulate) ------
  // 48 column-groups: 3 matrices x 16 N-tiles; each wave owns a column-group
  // and accumulates all 4 M-tiles against ONE set of B fragments per k-step,
  // so the 384 KB weight set is read ~once per workgroup.
  for (int cg = wave; cg < 48; cg += 8) {
    int mat = cg >> 4;              // 0=q 1=k 2=v
    int nt  = cg & 15;
    int nb  = nt * 16;
    const unsigned short* w = wt + (size_t)mat * D_ * D_;
    const float* bias = (mat == 0) ? bq : (mat == 1) ? bk : bv;
    float bval = bias[nb + (lane & 15)];
    v8f acc0, acc1, acc2, acc3;
#pragma unroll
    for (int r = 0; r < 8; ++r) { acc0[r] = bval; acc1[r] = bval; acc2[r] = bval; acc3[r] = bval; }
#pragma unroll
    for (int kb = 0; kb < D_; kb += 32) {
      v16bf bf = load_frag_B(w, nb, kb, lane);
      v16bf a0 = load_frag_A(sh_hn, 1 +  0, kb, lane);
      v16bf a1 = load_frag_A(sh_hn, 1 + 16, kb, lane);
      v16bf a2 = load_frag_A(sh_hn, 1 + 32, kb, lane);
      v16bf a3 = load_frag_A(sh_hn, 1 + 48, kb, lane);
      acc0 = wmma_bf16(a0, bf, acc0);
      acc1 = wmma_bf16(a1, bf, acc1);
      acc2 = wmma_bf16(a2, bf, acc2);
      acc3 = wmma_bf16(a3, bf, acc3);
    }
    unsigned short* dst = (mat == 0) ? sh_q : (mat == 1) ? sh_k : sh_v;
    int base = (mat == 0) ? 0 : 1;  // q indexed from t0; k/v carry the halo row
    int half = lane >> 4, n = nb + (lane & 15);
    store_tile(dst, base +  0, n, half, acc0);
    store_tile(dst, base + 16, n, half, acc1);
    store_tile(dst, base + 32, n, half, acc2);
    store_tile(dst, base + 48, n, half, acc3);
  }

  // ---- Phase B2: halo k/v rows (local 0 and 65) via GEMV ------------------
  for (int task = threadIdx.x; task < 1024; task += 256) {
    int c   = task & 255;
    int mat = (task >> 8) & 1;                 // 0=k 1=v
    int row = (task >> 9) ? (NR - 1) : 0;
    const unsigned short* w  = wt + (size_t)(1 + mat) * D_ * D_ + (size_t)c * D_;
    const unsigned short* hr = sh_hn + (size_t)row * LP;
    float acc = (mat ? bv : bk)[c];
    for (int kk = 0; kk < D_; ++kk) acc += bf2f(hr[kk]) * bf2f(w[kk]);
    (mat ? sh_v : sh_k)[(size_t)row * LP + c] = f2bf(acc);
  }
  __syncthreads();

  // ---- Phase C: banded attention (3 scores per row/head, softmax, mix) ----
  for (int task = threadIdx.x; task < TSZ * 8; task += 256) {
    int r  = task >> 3;      // 0..63 (t = t0 + r)
    int hh = task & 7;
    int t  = t0 + r;
    const unsigned short* qp  = sh_q + (size_t)r * LP + hh * DH_;
    const unsigned short* kpm = sh_k + (size_t)r * LP + hh * DH_;   // t-1
    const unsigned short* kp0 = kpm + LP;                           // t
    const unsigned short* kpp = kp0 + LP;                           // t+1
    float sm = 0.f, s0 = 0.f, sp = 0.f;
#pragma unroll
    for (int d = 0; d < DH_; ++d) {
      float qv = bf2f(qp[d]);
      sm += qv * bf2f(kpm[d]);
      s0 += qv * bf2f(kp0[d]);
      sp += qv * bf2f(kpp[d]);
    }
    const float scale = 0.17677669529663687f;   // 1/sqrt(32)
    sm *= scale; s0 *= scale; sp *= scale;
    if (t == 0)      sm = -1e9f;
    if (t == T_ - 1) sp = -1e9f;
    float mx = fmaxf(s0, fmaxf(sm, sp));
    float em = __expf(sm - mx), e0 = __expf(s0 - mx), ep = __expf(sp - mx);
    float inv = 1.f / (em + e0 + ep);
    em *= inv; e0 *= inv; ep *= inv;
    const unsigned short* vpm = sh_v + (size_t)r * LP + hh * DH_;
    const unsigned short* vp0 = vpm + LP;
    const unsigned short* vpp = vp0 + LP;
    unsigned short* ap = sh_a + (size_t)r * LP + hh * DH_;
#pragma unroll
    for (int d = 0; d < DH_; ++d)
      ap[d] = f2bf(em * bf2f(vpm[d]) + e0 * bf2f(vp0[d]) + ep * bf2f(vpp[d]));
  }
  __syncthreads();

  // ---- Phase D: h_out = h_in + attn @ Wo + bo via WMMA --------------------
  // Same B-fragment reuse: one N-tile per wave iteration, 4 M-tiles.
  const unsigned short* wo = wt + (size_t)3 * D_ * D_;
  for (int nt = wave; nt < 16; nt += 8) {
    int nb = nt * 16;
    float bval = bo[nb + (lane & 15)];
    v8f acc0, acc1, acc2, acc3;
#pragma unroll
    for (int r = 0; r < 8; ++r) { acc0[r] = bval; acc1[r] = bval; acc2[r] = bval; acc3[r] = bval; }
#pragma unroll
    for (int kb = 0; kb < D_; kb += 32) {
      v16bf bf = load_frag_B(wo, nb, kb, lane);
      v16bf a0 = load_frag_A(sh_a,  0, kb, lane);
      v16bf a1 = load_frag_A(sh_a, 16, kb, lane);
      v16bf a2 = load_frag_A(sh_a, 32, kb, lane);
      v16bf a3 = load_frag_A(sh_a, 48, kb, lane);
      acc0 = wmma_bf16(a0, bf, acc0);
      acc1 = wmma_bf16(a1, bf, acc1);
      acc2 = wmma_bf16(a2, bf, acc2);
      acc3 = wmma_bf16(a3, bf, acc3);
    }
    int half = lane >> 4, n = nb + (lane & 15);
#pragma unroll
    for (int mt = 0; mt < 4; ++mt) {
      v8f acc = (mt == 0) ? acc0 : (mt == 1) ? acc1 : (mt == 2) ? acc2 : acc3;
#pragma unroll
      for (int r = 0; r < 8; ++r) {
        int t = t0 + mt * 16 + r + 8 * half;
        size_t idx = ((size_t)b * T_ + t) * D_ + n;
        h_out[idx] = h_in[idx] + acc[r];
      }
    }
  }
}

// ---------------------------------------------------------------------------
// Final LayerNorm (in place on d_out is safe: wave reads row then writes it)
// ---------------------------------------------------------------------------
__global__ __launch_bounds__(256)
void final_ln(const float* __restrict__ h, const float* __restrict__ g,
              const float* __restrict__ bb, float* __restrict__ out) {
  int lane = threadIdx.x & 31, wave = threadIdx.x >> 5;
  size_t row = (size_t)blockIdx.x * 8 + wave;
  const float* rp = h + row * D_;
  float x[8], s = 0.f, s2 = 0.f;
#pragma unroll
  for (int j = 0; j < 8; ++j) {
    x[j] = rp[lane * 8 + j];
    s += x[j]; s2 += x[j] * x[j];
  }
#pragma unroll
  for (int m = 16; m >= 1; m >>= 1) {
    s  += __shfl_xor(s,  m, 32);
    s2 += __shfl_xor(s2, m, 32);
  }
  float mu   = s * (1.f / D_);
  float rstd = rsqrtf(s2 * (1.f / D_) - mu * mu + 1e-5f);
#pragma unroll
  for (int j = 0; j < 8; ++j) {
    int c = lane * 8 + j;
    out[row * D_ + c] = (x[j] - mu) * rstd * g[c] + bb[c];
  }
}

// ---------------------------------------------------------------------------
extern "C" void kernel_launch(void* const* d_in, const int* in_sizes, int n_in,
                              void* d_out, int out_size, void* d_ws, size_t ws_size,
                              hipStream_t stream) {
  const float* s    = (const float*)d_in[0];
  const float* W_in = (const float*)d_in[1];
  const float* b_in = (const float*)d_in[2];
  const float* Wq   = (const float*)d_in[3];
  const float* bq   = (const float*)d_in[4];
  const float* Wk   = (const float*)d_in[5];
  const float* bk   = (const float*)d_in[6];
  const float* Wv   = (const float*)d_in[7];
  const float* bv   = (const float*)d_in[8];
  const float* Wo   = (const float*)d_in[9];
  const float* bo   = (const float*)d_in[10];
  const float* ln_g = (const float*)d_in[11];
  const float* ln_b = (const float*)d_in[12];
  const float* fg   = (const float*)d_in[13];
  const float* fb   = (const float*)d_in[14];
  float* out = (float*)d_out;

  // Workspace: h ping buffer (f32) + transposed bf16 weights. d_out doubles
  // as the pong buffer (written fully by layers 0 and 2).
  float* h_a = (float*)d_ws;
  unsigned short* wt_all = (unsigned short*)(h_a + (size_t)BT * D_);
  float* h_b = out;

  convert_weights<<<(DEPTH_ * 4 * D_ * D_ + 255) / 256, 256, 0, stream>>>(
      Wq, Wk, Wv, Wo, wt_all);
  in_proj<<<(int)BT, 256, 0, stream>>>(s, W_in, b_in, h_a);

  dim3 grid(T_ / TSZ, B_);
  for (int l = 0; l < DEPTH_; ++l) {
    const float* hin = (l & 1) ? h_b : h_a;
    float* hout      = (l & 1) ? h_a : h_b;
    const unsigned short* wt = wt_all + (size_t)l * 4 * D_ * D_;
    layer_fused<<<grid, 256, 0, stream>>>(hin, hout, wt,
        bq + (size_t)l * D_, bk + (size_t)l * D_, bv + (size_t)l * D_,
        bo + (size_t)l * D_, ln_g + (size_t)l * D_, ln_b + (size_t)l * D_);
  }
  final_ln<<<(int)(BT / 8), 256, 0, stream>>>(h_b, fg, fb, out);
}
